// ABCAttention_30743375905059
// MI455X (gfx1250) — compile-verified, matrix operand order checked by reading, not run on hardware
//
#include <hip/hip_runtime.h>
#include <hip/hip_bf16.h>
#include <math.h>

// ---------------- problem constants (from reference) ----------------
#define TT   2048   // sequence length
#define DD   2048   // model dim
#define HH   16     // heads
#define DKK  64     // key dim per head
#define DVV  128    // value dim per head
#define MM   64     // slots per head
#define KEYD 1024   // H*DK
#define VALD 2048   // H*DV
#define SLOTD 1024  // H*M

typedef __attribute__((ext_vector_type(16))) __bf16 v16bf;
typedef __attribute__((ext_vector_type(8)))  float  v8f;

__device__ __forceinline__ __bf16 f2bf(float f) {
    union { float f; unsigned u; } x; x.f = f;
    unsigned r = x.u + 0x7FFFu + ((x.u >> 16) & 1u);   // round-to-nearest-even
    unsigned short h = (unsigned short)(r >> 16);
    union { unsigned short u; __bf16 b; } y; y.u = h;
    return y.b;
}

// ---------------- conversion kernels ----------------
__global__ void conv_f32_bf16(const float* __restrict__ in, __bf16* __restrict__ out, int n) {
    for (int i = blockIdx.x * blockDim.x + threadIdx.x; i < n; i += gridDim.x * blockDim.x)
        out[i] = f2bf(in[i]);
}

// Wt[n*K + k] = W[k*N + n]  (transpose + bf16), so B fragments are contiguous in K
__global__ void transpose_f32_bf16(const float* __restrict__ W, __bf16* __restrict__ Wt,
                                   int K, int N) {
    int total = K * N;
    for (int i = blockIdx.x * blockDim.x + threadIdx.x; i < total; i += gridDim.x * blockDim.x) {
        int k = i / N, n = i - k * N;
        Wt[(size_t)n * K + k] = f2bf(W[i]);
    }
}

// ---------------- BF16 WMMA GEMM, 32x64 register tile per wave ----------------
// C[M,N] = A[M,K] * B[K,N]; A row-major bf16, Bt = B^T row-major bf16 ([N,K]).
// Block: 256 threads = 8 waves arranged 4 (rows) x 2 (cols).
// Block tile: 128 rows x 128 cols; per-wave tile 32x64 = 2x4 WMMA fragments.
// 8 v_wmma per K-step of 32 vs 6 fragment loads; 64 accumulator VGPRs -> no spill.
__global__ void __launch_bounds__(256, 1)
gemm_bf16_wmma(const __bf16* __restrict__ A, const __bf16* __restrict__ Bt,
               float* __restrict__ C, int Mdim, int Ndim, int Kdim, int clampS) {
    const int wave = threadIdx.x >> 5;
    const int lane = threadIdx.x & 31;
    const int half = lane >> 4;          // which K-half of the fragment this lane holds
    const int l16  = lane & 15;
    const int waveRow = wave >> 1;       // 0..3
    const int waveCol = wave & 1;        // 0..1
    const int rowBase = blockIdx.y * 128 + waveRow * 32;
    const int colBase = blockIdx.x * 128 + waveCol * 64;

    v8f acc[2][4] = {};                  // [rowFrag][colFrag] = 64 accumulator VGPRs

    const __bf16* aBase = A  + (size_t)(rowBase + l16) * Kdim + half * 16;
    const __bf16* bBase = Bt + (size_t)(colBase + l16) * Kdim + half * 16;
    const size_t strideFrag = (size_t)16 * Kdim;   // 16 rows/cols down

    for (int k = 0; k < Kdim; k += 32) {
        v16bf afrag0 = *(const v16bf*)(aBase + k);
        v16bf afrag1 = *(const v16bf*)(aBase + strideFrag + k);
#pragma unroll
        for (int c = 0; c < 4; ++c) {
            v16bf bfrag = *(const v16bf*)(bBase + (size_t)c * strideFrag + k);
            acc[0][c] = __builtin_amdgcn_wmma_f32_16x16x32_bf16(
                false, afrag0, false, bfrag, (short)0, acc[0][c], false, false);
            acc[1][c] = __builtin_amdgcn_wmma_f32_16x16x32_bf16(
                false, afrag1, false, bfrag, (short)0, acc[1][c], false, false);
        }
    }

    // C/D layout: lane&15 = column, VGPR i -> row i + 8*(lane>>4)
#pragma unroll
    for (int r = 0; r < 2; ++r) {
#pragma unroll
        for (int c = 0; c < 4; ++c) {
            int col = colBase + c * 16 + l16;
#pragma unroll
            for (int i = 0; i < 8; ++i) {
                int row = rowBase + r * 16 + i + 8 * half;
                float v = acc[r][c][i];
                if (clampS) v = fminf(fmaxf(v, -32.0f), 32.0f);
                C[(size_t)row * Ndim + col] = v;
            }
        }
    }
}

// ---------------- sequential ABC recurrence, one block per head ----------------
// q,k,s: [T, H*64]; v: [T, H*128]; o: [H][T][128]
__global__ void abc_scan(const float* __restrict__ q, const float* __restrict__ k,
                         const float* __restrict__ v, const float* __restrict__ s,
                         float* __restrict__ o) {
    const int h = blockIdx.x;
    const int tid = threadIdx.x;
    __shared__ float hk[64 * 65];    // [k][m], padded
    __shared__ float hv[64 * 129];   // [m][dv], padded
    __shared__ float z[64], gg[64], pp[64], ok[64], pv[64];

    for (int i = tid; i < 64 * 65; i += 256) hk[i] = 0.0f;
    for (int i = tid; i < 64 * 129; i += 256) hv[i] = 0.0f;
    if (tid < 64) z[tid] = -INFINITY;
    __syncthreads();

    const float scale = 0.125f; // DK^-0.5

    for (int t = 0; t < TT; ++t) {
        const float* st = s + (size_t)t * SLOTD + h * MM;
        const float* qt = q + (size_t)t * KEYD + h * DKK;
        const float* kt = k + (size_t)t * KEYD + h * DKK;
        const float* vt = v + (size_t)t * VALD + h * DVV;

        if (tid < 64) {                         // cumlogsumexp -> decay g, write weight p
            float sv = st[tid];
            float zp = z[tid];
            float zn, gv;
            if (zp == -INFINITY) { zn = sv; gv = 1.0f; }
            else {
                float mx = fmaxf(zp, sv);
                zn = mx + logf(expf(zp - mx) + expf(sv - mx));
                gv = expf(zp - zn);
            }
            z[tid] = zn; gg[tid] = gv; pp[tid] = expf(sv - zn);
        }
        __syncthreads();

        for (int i = tid; i < 4096; i += 256) { // hk = hk*g[m] + k[kk]*p[m]
            int kk = i >> 6, m = i & 63;
            hk[kk * 65 + m] = hk[kk * 65 + m] * gg[m] + kt[kk] * pp[m];
        }
        __syncthreads();

        if (tid < 64) {                          // slot logits  ok[m] = (q*scale) . hk[:,m]
            float a = 0.0f;
            for (int kk = 0; kk < 64; ++kk) a += qt[kk] * hk[kk * 65 + tid];
            ok[tid] = a * scale;
        }
        __syncthreads();
        if (tid < 64) {                          // softmax over M=64
            float mx = ok[0];
            for (int m2 = 1; m2 < 64; ++m2) mx = fmaxf(mx, ok[m2]);
            float sum = 0.0f;
            for (int m2 = 0; m2 < 64; ++m2) sum += expf(ok[m2] - mx);
            pv[tid] = expf(ok[tid] - mx) / sum;
        }
        __syncthreads();

        for (int i = tid; i < 8192; i += 256) { // hv = hv*g[m] + p[m]*v[dv]
            int m = i >> 7, dv = i & 127;
            hv[m * 129 + dv] = hv[m * 129 + dv] * gg[m] + pp[m] * vt[dv];
        }
        __syncthreads();

        if (tid < 128) {                         // o[dv] = pv . hv[:,dv]
            float a = 0.0f;
            for (int m = 0; m < 64; ++m) a += pv[m] * hv[m * 129 + tid];
            o[((size_t)h * TT + t) * DVV + tid] = a;
        }
        __syncthreads();
    }
}

// ---------------- RMSNorm * weight * swish(g), one wave per (t,h) row ----------------
// o_att: [H][T][128]; g: [T][2048]; out (bf16): [T][2048]
__global__ void gate_kernel(const float* __restrict__ o_att, const float* __restrict__ g,
                            const float* __restrict__ w, __bf16* __restrict__ out) {
    int gid  = blockIdx.x * (blockDim.x >> 5) + (threadIdx.x >> 5);
    int lane = threadIdx.x & 31;
    int t = gid >> 4, h = gid & 15;
    const float* op = o_att + ((size_t)h * TT + t) * DVV;

    float vals[4];
    float ss = 0.0f;
#pragma unroll
    for (int i = 0; i < 4; ++i) {
        float x = op[lane * 4 + i];
        vals[i] = x; ss += x * x;
    }
#pragma unroll
    for (int off = 16; off > 0; off >>= 1) ss += __shfl_xor(ss, off, 32);
    float r = rsqrtf(ss * (1.0f / 128.0f) + 1e-5f);

    const float* gp = g + (size_t)t * VALD + h * DVV;
    __bf16* ob = out + (size_t)t * VALD + h * DVV;
#pragma unroll
    for (int i = 0; i < 4; ++i) {
        int dv = lane * 4 + i;
        float gv = gp[dv];
        float sw = gv / (1.0f + expf(-gv));          // swish
        ob[dv] = f2bf(vals[i] * r * w[dv] * sw);
    }
}

// ---------------- host-side orchestration ----------------
extern "C" void kernel_launch(void* const* d_in, const int* in_sizes, int n_in,
                              void* d_out, int out_size, void* d_ws, size_t ws_size,
                              hipStream_t stream) {
    const float* hs = (const float*)d_in[0];   // [2048,2048]
    const float* Wq = (const float*)d_in[1];   // [2048,1024]
    const float* Wk = (const float*)d_in[2];   // [2048,1024]
    const float* Wv = (const float*)d_in[3];   // [2048,2048]
    const float* Wg = (const float*)d_in[4];   // [2048,2048]
    const float* Ws = (const float*)d_in[5];   // [2048,1024]
    const float* Wo = (const float*)d_in[6];   // [2048,2048]
    const float* gw = (const float*)d_in[7];   // [128]

    // workspace carve-up
    char* p = (char*)d_ws;
    auto take = [&](size_t bytes) { char* r = p; p += (bytes + 255) & ~(size_t)255; return r; };
    __bf16* Xb   = (__bf16*)take((size_t)TT * DD * 2);
    __bf16* WqT  = (__bf16*)take((size_t)DD * KEYD * 2);
    __bf16* WkT  = (__bf16*)take((size_t)DD * KEYD * 2);
    __bf16* WvT  = (__bf16*)take((size_t)DD * VALD * 2);
    __bf16* WgT  = (__bf16*)take((size_t)DD * VALD * 2);
    __bf16* WsT  = (__bf16*)take((size_t)DD * SLOTD * 2);
    __bf16* WoT  = (__bf16*)take((size_t)VALD * DD * 2);
    float*  qf   = (float*)take((size_t)TT * KEYD * 4);
    float*  kf   = (float*)take((size_t)TT * KEYD * 4);
    float*  sf   = (float*)take((size_t)TT * SLOTD * 4);
    float*  vf   = (float*)take((size_t)TT * VALD * 4);
    float*  gf   = (float*)take((size_t)TT * VALD * 4);
    float*  oatt = (float*)take((size_t)HH * TT * DVV * 4);
    __bf16* gated= (__bf16*)take((size_t)TT * VALD * 2);

    // 1) precision conversion (X) + weight transpose+convert
    conv_f32_bf16<<<4096, 256, 0, stream>>>(hs, Xb, TT * DD);
    transpose_f32_bf16<<<4096, 256, 0, stream>>>(Wq, WqT, DD, KEYD);
    transpose_f32_bf16<<<4096, 256, 0, stream>>>(Wk, WkT, DD, KEYD);
    transpose_f32_bf16<<<4096, 256, 0, stream>>>(Wv, WvT, DD, VALD);
    transpose_f32_bf16<<<4096, 256, 0, stream>>>(Wg, WgT, DD, VALD);
    transpose_f32_bf16<<<4096, 256, 0, stream>>>(Ws, WsT, DD, SLOTD);
    transpose_f32_bf16<<<4096, 256, 0, stream>>>(Wo, WoT, VALD, DD);

    // 2) projection GEMMs (WMMA bf16 -> f32); block tile 128x128
    dim3 blk(256);
    gemm_bf16_wmma<<<dim3(KEYD / 128, TT / 128), blk, 0, stream>>>(Xb, WqT, qf, TT, KEYD, DD, 0);
    gemm_bf16_wmma<<<dim3(KEYD / 128, TT / 128), blk, 0, stream>>>(Xb, WkT, kf, TT, KEYD, DD, 0);
    gemm_bf16_wmma<<<dim3(VALD / 128, TT / 128), blk, 0, stream>>>(Xb, WvT, vf, TT, VALD, DD, 0);
    gemm_bf16_wmma<<<dim3(VALD / 128, TT / 128), blk, 0, stream>>>(Xb, WgT, gf, TT, VALD, DD, 0);
    gemm_bf16_wmma<<<dim3(SLOTD / 128, TT / 128), blk, 0, stream>>>(Xb, WsT, sf, TT, SLOTD, DD, 1);

    // 3) sequential ABC recurrence (one WGP workgroup per head, state in LDS)
    abc_scan<<<HH, 256, 0, stream>>>(qf, kf, vf, sf, oatt);

    // 4) fused RMSNorm * weight * swish(g), emit bf16 A-operand for final GEMM
    gate_kernel<<<(TT * HH) / 8, 256, 0, stream>>>(oatt, gf, gw, gated);

    // 5) output projection GEMM -> d_out (f32)
    gemm_bf16_wmma<<<dim3(DD / 128, TT / 128), blk, 0, stream>>>(gated, WoT, (float*)d_out,
                                                                 TT, DD, VALD, 0);
}
